// PointNet2SemSegSSG_64768106823695
// MI455X (gfx1250) — matrix operand order repaired and checked by reading, not compiled, not run on hardware
//
#include <hip/hip_runtime.h>

typedef __attribute__((ext_vector_type(16))) _Float16 v16h;
typedef __attribute__((ext_vector_type(8)))  _Float16 v8h;
typedef __attribute__((ext_vector_type(8)))  float    v8f;

#define BN_RS 0.99999500003749968f   // 1/sqrt(1+1e-5)
#define NB 16
#define NP 1024

__constant__ int c_PATCH_UP[21]   = {17,2,3,4,4,6,7,8,8,10,11,12,12,14,15,16,16,18,19,20,20};
__constant__ int c_PATCH_DOWN[21] = {0,1,1,2,3,5,5,6,7,9,9,10,11,13,13,14,15,0,17,18,19};

// ---------------------------------------------------------------------------
// One-shot: convert weight matrix (Cout x Cin, f32) -> (Cout x Kpad, f16),
// zero-padding K. Removes all guards/converts from the WMMA hot loops.
// ---------------------------------------------------------------------------
__global__ void weight_cvt_kernel(const float* __restrict__ W, int Cin, int Kpad,
                                  _Float16* __restrict__ W16, int total) {
  int t = blockIdx.x * blockDim.x + threadIdx.x;
  if (t >= total) return;
  int o = t / Kpad, k = t % Kpad;
  W16[t] = (k < Cin) ? (_Float16)W[(size_t)o * Cin + k] : (_Float16)0.f;
}

// ---------------------------------------------------------------------------
// Farthest point sampling: one block per batch, 256 threads, LDS argmax.
// ---------------------------------------------------------------------------
__global__ void fps_kernel(const float* __restrict__ pts, int n, int stride,
                           int npoint, int* __restrict__ fidx) {
  int b = blockIdx.x;
  const float* P = pts + (size_t)b * n * stride;
  int* out = fidx + (size_t)b * npoint;
  __shared__ float rv[256];
  __shared__ int   ri[256];
  __shared__ int   s_last;
  float dist[4];
  int per = n / 256;
  for (int i = 0; i < per; ++i) dist[i] = 1e10f;
  if (threadIdx.x == 0) { out[0] = 0; s_last = 0; }
  __syncthreads();
  for (int it = 1; it < npoint; ++it) {
    int last = s_last;
    float lx = P[last*stride], ly = P[last*stride+1], lz = P[last*stride+2];
    float best = -1.f; int bidx = 0;
    for (int i = 0; i < per; ++i) {
      int p = threadIdx.x + i * 256;
      float dx = P[p*stride] - lx, dy = P[p*stride+1] - ly, dz = P[p*stride+2] - lz;
      float d = dx*dx + dy*dy + dz*dz;
      dist[i] = fminf(dist[i], d);
      if (dist[i] > best || (dist[i] == best && p < bidx)) { best = dist[i]; bidx = p; }
    }
    rv[threadIdx.x] = best; ri[threadIdx.x] = bidx;
    __syncthreads();
    for (int s = 128; s > 0; s >>= 1) {
      if (threadIdx.x < s) {
        float v2 = rv[threadIdx.x + s]; int i2 = ri[threadIdx.x + s];
        if (v2 > rv[threadIdx.x] || (v2 == rv[threadIdx.x] && i2 < ri[threadIdx.x])) {
          rv[threadIdx.x] = v2; ri[threadIdx.x] = i2;
        }
      }
      __syncthreads();
    }
    if (threadIdx.x == 0) { out[it] = ri[0]; s_last = ri[0]; }
    __syncthreads();
  }
}

// ---------------------------------------------------------------------------
// Ball-query + group: one wave per center (4 waves / block).
// ---------------------------------------------------------------------------
__global__ void group_kernel(const float* __restrict__ xyz, int xs, long xb,
                             const float* __restrict__ feats, int fp, int fcs, long fb, int Cf,
                             const int* __restrict__ fidx, int M, int Npts, float radius,
                             _Float16* __restrict__ grouped, float* __restrict__ centers) {
  int w = threadIdx.x >> 5;
  int lane = threadIdx.x & 31;
  int cidx = blockIdx.x * 4 + w;
  int b = cidx / M, m = cidx % M;
  __shared__ int sidx[4][64];
  const float* X = xyz + (size_t)b * xb;
  int ci = fidx[b * M + m];
  float cx = X[ci*xs], cy = X[ci*xs+1], cz = X[ci*xs+2];
  if (lane == 0) {
    centers[(size_t)(b*M+m)*3+0] = cx;
    centers[(size_t)(b*M+m)*3+1] = cy;
    centers[(size_t)(b*M+m)*3+2] = cz;
    float r2 = radius * radius;
    int cnt = 0;
    for (int j = 0; j < Npts && cnt < 64; ++j) {
      float dx = X[j*xs]-cx, dy = X[j*xs+1]-cy, dz = X[j*xs+2]-cz;
      if (dx*dx + dy*dy + dz*dz < r2) sidx[w][cnt++] = j;
    }
    int f0 = cnt ? sidx[w][0] : 0;
    for (; cnt < 64; ++cnt) sidx[w][cnt] = f0;
  }
  __syncthreads();
  const float* F = feats + (size_t)b * fb;
  int CC = 3 + Cf;
  _Float16* G = grouped + (size_t)(b*M+m) * 64 * CC;
  for (int s = lane; s < 64; s += 32) {
    int j = sidx[w][s];
    G[s*CC+0] = (_Float16)(X[j*xs]   - cx);
    G[s*CC+1] = (_Float16)(X[j*xs+1] - cy);
    G[s*CC+2] = (_Float16)(X[j*xs+2] - cz);
    for (int c = 0; c < Cf; ++c)
      G[s*CC+3+c] = (_Float16)F[(size_t)j*fp + (size_t)c*fcs];
  }
}

// ---------------------------------------------------------------------------
// Refine-branch ball-query: centers = estimated joints, raw 6-ch gather.
// ---------------------------------------------------------------------------
__global__ void refine_group_kernel(const float* __restrict__ pc,
                                    const float* __restrict__ est,
                                    _Float16* __restrict__ rnpc) {
  int w = threadIdx.x >> 5;
  int lane = threadIdx.x & 31;
  int cidx = blockIdx.x * 4 + w;
  int b = cidx / 21, m = cidx % 21;
  __shared__ int sidx[4][64];
  const float* P = pc + (size_t)b * NP * 6;
  float cx = est[(b*21+m)*3], cy = est[(b*21+m)*3+1], cz = est[(b*21+m)*3+2];
  if (lane == 0) {
    int cnt = 0;
    for (int j = 0; j < NP && cnt < 64; ++j) {
      float dx = P[j*6]-cx, dy = P[j*6+1]-cy, dz = P[j*6+2]-cz;
      if (dx*dx + dy*dy + dz*dz < 0.01f) sidx[w][cnt++] = j;
    }
    int f0 = cnt ? sidx[w][0] : 0;
    for (; cnt < 64; ++cnt) sidx[w][cnt] = f0;
  }
  __syncthreads();
  for (int s = lane; s < 64; s += 32) {
    int j = sidx[w][s];
    _Float16* D = rnpc + ((size_t)(b*21+m)*64 + s) * 6;
    for (int c = 0; c < 6; ++c) D[c] = (_Float16)P[j*6+c];
  }
}

__global__ void patch_expand_kernel(const _Float16* __restrict__ rnpc,
                                    _Float16* __restrict__ rg) {
  int t = blockIdx.x * blockDim.x + threadIdx.x;
  if (t >= NB*21*192) return;
  int s = t % 192, m = (t/192) % 21, b = t/(192*21);
  int sm = (s < 64) ? c_PATCH_UP[m] : ((s < 128) ? m : c_PATCH_DOWN[m]);
  int ss = s & 63;
  const _Float16* S = rnpc + ((size_t)(b*21+sm)*64 + ss) * 6;
  _Float16* D = rg + (size_t)t * 6;
  for (int c = 0; c < 6; ++c) D[c] = S[c];
}

// ---------------------------------------------------------------------------
// Fragment loaders: A from LDS (two contiguous 8-half chunks per lane),
// B from pre-padded f16 weights (two contiguous 8-half chunks per lane).
// ---------------------------------------------------------------------------
__device__ inline v16h load_a_frag(const _Float16* A, int rowBase, int k0, int lh) {
  v8h alo = *(const v8h*)(A + rowBase + k0 + lh*8);
  v8h ahi = *(const v8h*)(A + rowBase + k0 + 16 + lh*8);
  return __builtin_shufflevector(alo, ahi, 0,1,2,3,4,5,6,7,8,9,10,11,12,13,14,15);
}
__device__ inline v16h load_b_frag(const _Float16* wrow, int k0, int lh) {
  v8h blo = *(const v8h*)(wrow + k0 + lh*8);
  v8h bhi = *(const v8h*)(wrow + k0 + 16 + lh*8);
  return __builtin_shufflevector(blo, bhi, 0,1,2,3,4,5,6,7,8,9,10,11,12,13,14,15);
}

// ---------------------------------------------------------------------------
// WMMA layer: 16x16 output tile per wave, K in steps of 32 (f16, f32 acc).
// A staged in LDS (row-major, stride Kpad); W16 pre-padded (Cout x Kpad, f16).
// ---------------------------------------------------------------------------
__device__ inline void wmma_layer_lds(const _Float16* A, int Kpad,
                                      const _Float16* __restrict__ W16,
                                      const float* __restrict__ bias,
                                      const float* __restrict__ g, const float* __restrict__ beta,
                                      _Float16* Y, int Cout, int ROWS) {
  int wave = threadIdx.x >> 5, nw = blockDim.x >> 5;
  int lane = threadIdx.x & 31, lh = lane >> 4, mn = lane & 15;
  int rt = ROWS >> 4, ct = Cout >> 4;
  for (int t = wave; t < rt * ct; t += nw) {
    int r0 = (t % rt) << 4, n0 = (t / rt) << 4;
    v8f acc = {0.f,0.f,0.f,0.f,0.f,0.f,0.f,0.f};
    const _Float16* wrow = W16 + (size_t)(n0 + mn) * Kpad;
    int rowBase = (r0 + mn) * Kpad;
    for (int k0 = 0; k0 < Kpad; k0 += 32) {
      v16h av = load_a_frag(A, rowBase, k0, lh);
      v16h bv = load_b_frag(wrow, k0, lh);
      acc = __builtin_amdgcn_wmma_f32_16x16x32_f16(false, av, false, bv, (short)0, acc, false, false);
    }
    int n = n0 + mn;
    float bs = bias ? bias[n] : 0.f;
    float sc = g[n] * BN_RS, sh = beta[n];
#pragma unroll
    for (int j = 0; j < 8; ++j) {
      float v = fmaxf((acc[j] + bs) * sc + sh, 0.f);
      Y[(r0 + lh*8 + j) * Cout + n] = (_Float16)v;
    }
  }
}

// Last layer: ReLU then max over rows, folded via lane shuffle + uint atomicMax
// (valid: ReLU outputs are non-negative, so float order == uint order).
__device__ inline void wmma_layer_max(const _Float16* A, int Kpad,
                                      const _Float16* __restrict__ W16,
                                      const float* __restrict__ bias,
                                      const float* __restrict__ g, const float* __restrict__ beta,
                                      float* __restrict__ out, int Cout, int ROWS) {
  int wave = threadIdx.x >> 5, nw = blockDim.x >> 5;
  int lane = threadIdx.x & 31, lh = lane >> 4, mn = lane & 15;
  int rt = ROWS >> 4, ct = Cout >> 4;
  for (int t = wave; t < rt * ct; t += nw) {
    int r0 = (t % rt) << 4, n0 = (t / rt) << 4;
    v8f acc = {0.f,0.f,0.f,0.f,0.f,0.f,0.f,0.f};
    const _Float16* wrow = W16 + (size_t)(n0 + mn) * Kpad;
    int rowBase = (r0 + mn) * Kpad;
    for (int k0 = 0; k0 < Kpad; k0 += 32) {
      v16h av = load_a_frag(A, rowBase, k0, lh);
      v16h bv = load_b_frag(wrow, k0, lh);
      acc = __builtin_amdgcn_wmma_f32_16x16x32_f16(false, av, false, bv, (short)0, acc, false, false);
    }
    int n = n0 + mn;
    float bs = bias ? bias[n] : 0.f;
    float sc = g[n] * BN_RS, sh = beta[n];
    float vmax = 0.f;
#pragma unroll
    for (int j = 0; j < 8; ++j)
      vmax = fmaxf(vmax, fmaxf((acc[j] + bs) * sc + sh, 0.f));
    vmax = fmaxf(vmax, __shfl_xor(vmax, 16, 32));
    if (lh == 0) atomicMax((unsigned int*)(out + n), __float_as_uint(vmax));
  }
}

// ---------------------------------------------------------------------------
// Fused 3-layer shared-MLP + sample-max. One block per ROWS-chunk of a group.
// All intermediates stay in LDS (f16).
// ---------------------------------------------------------------------------
template <int CIN, int C1, int C2, int C3, int ROWS>
__global__ void fused_mlp3_max(const _Float16* __restrict__ in, int S,
                               const _Float16* W1, const float* bias1, const float* g1, const float* b1,
                               const _Float16* W2, const float* bias2, const float* g2, const float* b2,
                               const _Float16* W3, const float* bias3, const float* g3, const float* b3,
                               float* __restrict__ out) {
  constexpr int K0 = (CIN + 31) & ~31;
  extern __shared__ _Float16 lds[];
  _Float16* a0 = lds;
  _Float16* a1 = a0 + ROWS * K0;
  _Float16* a2 = a1 + ROWS * C1;
  int chunks = S / ROWS;
  int grp = blockIdx.x / chunks;
  int ch  = blockIdx.x % chunks;
  const _Float16* src = in + ((size_t)grp * S + (size_t)ch * ROWS) * CIN;
  for (int i = threadIdx.x; i < ROWS * K0; i += blockDim.x) {
    int r = i / K0, c = i % K0;
    a0[i] = (c < CIN) ? src[r*CIN + c] : (_Float16)0.f;
  }
  __syncthreads();
  wmma_layer_lds(a0, K0, W1, bias1, g1, b1, a1, C1, ROWS);
  __syncthreads();
  wmma_layer_lds(a1, C1, W2, bias2, g2, b2, a2, C2, ROWS);
  __syncthreads();
  wmma_layer_max(a2, C2, W3, bias3, g3, b3, out + (size_t)grp * C3, C3, ROWS);
}

// ---------------------------------------------------------------------------
// Single-layer WMMA linear (+bias/BN/ReLU), f32 in/out. 16 rows per block.
// ---------------------------------------------------------------------------
__global__ void linear_kernel(const float* __restrict__ in, int Cin,
                              const _Float16* __restrict__ W16, const float* bias,
                              const float* scale, const float* shift,
                              float* __restrict__ out, int Cout, int relu) {
  int Kpad = (Cin + 31) & ~31;
  extern __shared__ _Float16 a0[];
  int r0 = blockIdx.x * 16;
  for (int i = threadIdx.x; i < 16 * Kpad; i += blockDim.x) {
    int r = i / Kpad, c = i % Kpad;
    a0[i] = (c < Cin) ? (_Float16)in[(size_t)(r0 + r) * Cin + c] : (_Float16)0.f;
  }
  __syncthreads();
  int wave = threadIdx.x >> 5, nw = blockDim.x >> 5;
  int lane = threadIdx.x & 31, lh = lane >> 4, mn = lane & 15;
  for (int tc = wave; tc < (Cout >> 4); tc += nw) {
    int n0 = tc << 4;
    v8f acc = {0.f,0.f,0.f,0.f,0.f,0.f,0.f,0.f};
    const _Float16* wrow = W16 + (size_t)(n0 + mn) * Kpad;
    int rowBase = mn * Kpad;
    for (int k0 = 0; k0 < Kpad; k0 += 32) {
      v16h av = load_a_frag(a0, rowBase, k0, lh);
      v16h bv = load_b_frag(wrow, k0, lh);
      acc = __builtin_amdgcn_wmma_f32_16x16x32_f16(false, av, false, bv, (short)0, acc, false, false);
    }
    int n = n0 + mn;
    float bs = bias ? bias[n] : 0.f;
    float sc = scale ? scale[n] * BN_RS : 1.f;
    float sh = shift ? shift[n] : 0.f;
#pragma unroll
    for (int j = 0; j < 8; ++j) {
      float v = (acc[j] + bs) * sc + sh;
      if (relu) v = fmaxf(v, 0.f);
      out[(size_t)(r0 + lh*8 + j) * Cout + n] = v;
    }
  }
}

// ---------------------------------------------------------------------------
// Glue / tail kernels (tiny, VALU).
// ---------------------------------------------------------------------------
__global__ void sa3_build_kernel(const float* __restrict__ c2, const float* __restrict__ l2f,
                                 _Float16* __restrict__ out) {
  int t = blockIdx.x * blockDim.x + threadIdx.x;
  if (t >= NB*128*259) return;
  int row = t / 259, c = t % 259;
  float v = (c < 3) ? c2[row*3 + c] : l2f[(size_t)row*256 + (c-3)];
  out[t] = (_Float16)v;
}

__global__ void fc3_kernel(const float* __restrict__ h2, const float* __restrict__ W3,
                           const float* __restrict__ b3, float* __restrict__ est) {
  int t = blockIdx.x * blockDim.x + threadIdx.x;
  if (t >= NB*63) return;
  int b = t / 63, o = t % 63;
  float s = 0.f;
  for (int k = 0; k < 512; ++k) s += h2[b*512 + k] * W3[o*512 + k];
  est[t] = s + b3[o];
}

__global__ void rfc_build_kernel(const float* __restrict__ est, const float* __restrict__ rfeat,
                                 float* __restrict__ out) {
  int t = blockIdx.x * blockDim.x + threadIdx.x;
  if (t >= NB*21*259) return;
  int row = t / 259, c = t % 259;
  out[t] = (c < 3) ? est[row*3 + c] : rfeat[(size_t)row*256 + (c-3)];
}

__global__ void refine_final_kernel(const float* __restrict__ h, const float* __restrict__ Wl,
                                    const float* __restrict__ bl, float* __restrict__ out) {
  int t = blockIdx.x * blockDim.x + threadIdx.x;
  if (t >= NB*21*3) return;
  int row = t / 3, c = t % 3;
  float s = 0.f;
  for (int k = 0; k < 128; ++k) s += h[row*128 + k] * Wl[c*128 + k];
  out[t] = s + bl[c];
}

// ---------------------------------------------------------------------------
extern "C" void kernel_launch(void* const* d_in, const int* in_sizes, int n_in,
                              void* d_out, int out_size, void* d_ws, size_t ws_size,
                              hipStream_t stream) {
  (void)in_sizes; (void)n_in; (void)out_size; (void)ws_size;
  auto IN = [&](int i) { return (const float*)d_in[i]; };
  const float* pc = IN(0);
  // sa1 (W,g,b)x3 : 1..9 ; sa2 : 10..18 ; sa3 : 19..27
  // fc : 28 W1,29 g1,30 b1,31 W2,32 g2,33 b2,34 W3,35 b3
  // rsa (W,bias,g,b)x3 : 36..47 ; rfc : 48..59 ; 60 Wl, 61 bl
  char* ws = (char*)d_ws;
  size_t off = 0;
  auto ALLOC = [&](size_t bytes) { size_t r = off; off = (off + bytes + 255) & ~(size_t)255; return r; };
  int*      fidx1    = (int*)     (ws + ALLOC((size_t)NB*512*4));
  float*    centers1 = (float*)   (ws + ALLOC((size_t)NB*512*3*4));
  _Float16* g1buf    = (_Float16*)(ws + ALLOC((size_t)NB*512*64*6*2));
  float*    l1f      = (float*)   (ws + ALLOC((size_t)NB*512*128*4));
  int*      fidx2    = (int*)     (ws + ALLOC((size_t)NB*128*4));
  float*    centers2 = (float*)   (ws + ALLOC((size_t)NB*128*3*4));
  _Float16* g2buf    = (_Float16*)(ws + ALLOC((size_t)NB*128*64*131*2));
  float*    l2f      = (float*)   (ws + ALLOC((size_t)NB*128*256*4));
  _Float16* sa3in    = (_Float16*)(ws + ALLOC((size_t)NB*128*259*2));
  float*    l3       = (float*)   (ws + ALLOC((size_t)NB*1024*4));
  float*    h1       = (float*)   (ws + ALLOC((size_t)NB*1024*4));
  float*    h2       = (float*)   (ws + ALLOC((size_t)NB*512*4));
  _Float16* rnpc     = (_Float16*)(ws + ALLOC((size_t)NB*21*64*6*2));
  _Float16* rg       = (_Float16*)(ws + ALLOC((size_t)NB*21*192*6*2));
  float*    rfeat    = (float*)   (ws + ALLOC((size_t)NB*21*256*4));
  float*    rfcin    = (float*)   (ws + ALLOC((size_t)NB*21*259*4));
  float*    rh1      = (float*)   (ws + ALLOC((size_t)NB*21*256*4));
  float*    rh2      = (float*)   (ws + ALLOC((size_t)NB*21*128*4));
  float*    rh3      = (float*)   (ws + ALLOC((size_t)NB*21*128*4));
  float* est    = (float*)d_out;          // (16,21,3) = (16,63)
  float* refine = (float*)d_out + NB*63;  // (16,21,3)

  // ---- Pre-convert all WMMA weight matrices to padded f16 ----
  auto CVT = [&](const float* W, int Cout, int Cin) -> const _Float16* {
    int Kpad = (Cin + 31) & ~31;
    int total = Cout * Kpad;
    _Float16* W16 = (_Float16*)(ws + ALLOC((size_t)total * 2));
    weight_cvt_kernel<<<(total + 255)/256, 256, 0, stream>>>(W, Cin, Kpad, W16, total);
    return W16;
  };
  const _Float16* w_s1a = CVT(IN(1),   64,    6);
  const _Float16* w_s1b = CVT(IN(4),   64,   64);
  const _Float16* w_s1c = CVT(IN(7),  128,   64);
  const _Float16* w_s2a = CVT(IN(10), 128,  131);
  const _Float16* w_s2b = CVT(IN(13), 128,  128);
  const _Float16* w_s2c = CVT(IN(16), 256,  128);
  const _Float16* w_s3a = CVT(IN(19), 256,  259);
  const _Float16* w_s3b = CVT(IN(22), 512,  256);
  const _Float16* w_s3c = CVT(IN(25), 1024, 512);
  const _Float16* w_fc1 = CVT(IN(28), 1024, 1024);
  const _Float16* w_fc2 = CVT(IN(31), 512,  1024);
  const _Float16* w_ra  = CVT(IN(36),  64,    6);
  const _Float16* w_rb  = CVT(IN(40), 128,   64);
  const _Float16* w_rc  = CVT(IN(44), 256,  128);
  const _Float16* w_f1  = CVT(IN(48), 256,  259);
  const _Float16* w_f2  = CVT(IN(52), 128,  256);
  const _Float16* w_f3  = CVT(IN(56), 128,  128);

  // ---- SA1 ----
  fps_kernel<<<NB, 256, 0, stream>>>(pc, NP, 6, 512, fidx1);
  group_kernel<<<(NB*512)/4, 128, 0, stream>>>(
      pc, 6, (long)NP*6, pc + 3, 6, 1, (long)NP*6, 3,
      fidx1, 512, NP, 0.1f, g1buf, centers1);
  hipMemsetAsync(l1f, 0, (size_t)NB*512*128*4, stream);
  fused_mlp3_max<6,64,64,128,64><<<NB*512, 256, (64*32+64*64+64*64)*2, stream>>>(
      g1buf, 64,
      w_s1a, nullptr, IN(2), IN(3),
      w_s1b, nullptr, IN(5), IN(6),
      w_s1c, nullptr, IN(8), IN(9), l1f);

  // ---- SA2 ----
  fps_kernel<<<NB, 256, 0, stream>>>(centers1, 512, 3, 128, fidx2);
  group_kernel<<<(NB*128)/4, 128, 0, stream>>>(
      centers1, 3, (long)512*3, l1f, 128, 1, (long)512*128, 128,
      fidx2, 128, 512, 0.2f, g2buf, centers2);
  hipMemsetAsync(l2f, 0, (size_t)NB*128*256*4, stream);
  fused_mlp3_max<131,128,128,256,64><<<NB*128, 256, (64*160+64*128+64*128)*2, stream>>>(
      g2buf, 64,
      w_s2a, nullptr, IN(11), IN(12),
      w_s2b, nullptr, IN(14), IN(15),
      w_s2c, nullptr, IN(17), IN(18), l2f);

  // ---- SA3 (global) ----
  sa3_build_kernel<<<(NB*128*259 + 255)/256, 256, 0, stream>>>(centers2, l2f, sa3in);
  hipMemsetAsync(l3, 0, (size_t)NB*1024*4, stream);
  fused_mlp3_max<259,256,512,1024,16><<<NB*8, 256, (16*288+16*256+16*512)*2, stream>>>(
      sa3in, 128,
      w_s3a, nullptr, IN(20), IN(21),
      w_s3b, nullptr, IN(23), IN(24),
      w_s3c, nullptr, IN(26), IN(27), l3);

  // ---- FC head -> est_centroids ----
  linear_kernel<<<NB/16, 256, 16*1024*2, stream>>>(l3, 1024, w_fc1, nullptr, IN(29), IN(30), h1, 1024, 1);
  linear_kernel<<<NB/16, 256, 16*1024*2, stream>>>(h1, 1024, w_fc2, nullptr, IN(32), IN(33), h2, 512, 1);
  fc3_kernel<<<(NB*63 + 255)/256, 256, 0, stream>>>(h2, IN(34), IN(35), est);

  // ---- Refine branch ----
  refine_group_kernel<<<(NB*21)/4, 128, 0, stream>>>(pc, est, rnpc);
  patch_expand_kernel<<<(NB*21*192 + 255)/256, 256, 0, stream>>>(rnpc, rg);
  hipMemsetAsync(rfeat, 0, (size_t)NB*21*256*4, stream);
  fused_mlp3_max<6,64,128,256,64><<<NB*21*3, 256, (64*32+64*64+64*128)*2, stream>>>(
      rg, 192,
      w_ra, IN(37), IN(38), IN(39),
      w_rb, IN(41), IN(42), IN(43),
      w_rc, IN(45), IN(46), IN(47), rfeat);
  rfc_build_kernel<<<(NB*21*259 + 255)/256, 256, 0, stream>>>(est, rfeat, rfcin);
  linear_kernel<<<(NB*21)/16, 256, 16*288*2, stream>>>(rfcin, 259, w_f1, IN(49), IN(50), IN(51), rh1, 256, 1);
  linear_kernel<<<(NB*21)/16, 256, 16*256*2, stream>>>(rh1, 256, w_f2, IN(53), IN(54), IN(55), rh2, 128, 1);
  linear_kernel<<<(NB*21)/16, 256, 16*128*2, stream>>>(rh2, 128, w_f3, IN(57), IN(58), IN(59), rh3, 128, 1);
  refine_final_kernel<<<(NB*21*3 + 255)/256, 256, 0, stream>>>(rh3, IN(60), IN(61), refine);
}
